// YOLOv3Head_16578573762645
// MI455X (gfx1250) — compile-verified, matrix operand order checked by reading, not run on hardware
//
#include <hip/hip_runtime.h>
#include <hip/hip_bf16.h>
#include <math.h>

#define BATCH   8
#define NUMC    80
#define NTOT    25200      // 19200 + 4800 + 1200
#define KTOP    300
#define KPAD    304        // 19 * 16
#define NTILE   19
#define MAXDET  300
#define CONF_T  0.25f
#define IOU_T   0.45f

typedef float    v2f  __attribute__((ext_vector_type(2)));
typedef float    v8f  __attribute__((ext_vector_type(8)));
typedef _Float16 v16h __attribute__((ext_vector_type(16)));

#ifdef __has_builtin
#if __has_builtin(__builtin_amdgcn_wmma_f32_16x16x4_f32)
#define HAVE_WMMA_F32X4 1
#endif
#if __has_builtin(__builtin_amdgcn_global_load_async_to_lds_b128) && \
    __has_builtin(__builtin_amdgcn_s_wait_asynccnt)
#define HAVE_ASYNC_LDS 1
#endif
#endif

#ifdef HAVE_ASYNC_LDS
// b128 payload type per clang diagnostic: GCC-style int 4-vector.
typedef int v4i_vs __attribute__((vector_size(16)));
typedef __attribute__((address_space(1))) v4i_vs GV4;   // __device__ (global)
typedef __attribute__((address_space(3))) v4i_vs LV4;   // __shared__ (LDS)
// AS1 address == generic global address; AS3 offset == low 32 bits of generic LDS addr.
__device__ __forceinline__ GV4* to_global4(const void* p) {
    return (GV4*)(unsigned long long)p;
}
__device__ __forceinline__ LV4* to_lds4(const void* p) {
    return (LV4*)(unsigned int)(unsigned long long)p;
}
// Copy nvec4 16-byte chunks global->LDS with async-to-LDS ops, then drain ASYNCcnt.
__device__ __forceinline__ void async_stage(const float* __restrict__ src,
                                            float* __restrict__ dst_lds,
                                            int nvec4, int tid, int nthreads) {
    for (int v = tid; v < nvec4; v += nthreads)
        __builtin_amdgcn_global_load_async_to_lds_b128(to_global4(src + v * 4),
                                                       to_lds4(dst_lds + v * 4), 0, 0);
    __builtin_amdgcn_s_wait_asynccnt(0);
}
#endif

__constant__ float c_ancw[9] = {10.f,16.f,33.f, 30.f,62.f,59.f, 116.f,156.f,373.f};
__constant__ float c_anch[9] = {13.f,30.f,23.f, 61.f,45.f,119.f, 90.f,198.f,326.f};

// v_rcp_f32-based sigmoid: avoids the ~10-op IEEE divide sequence.
__device__ __forceinline__ float sigmoidf_(float x) {
    return __builtin_amdgcn_rcpf(1.0f + __expf(-x));
}

// ---------------------------------------------------------------------------
// Kernel 1: decode all three scales -> boxes [B,N,4], scores [B,C,N]
// ---------------------------------------------------------------------------
__global__ void yolo_decode(const float* __restrict__ p0,
                            const float* __restrict__ p1,
                            const float* __restrict__ p2,
                            float* __restrict__ boxes,
                            float* __restrict__ scores)
{
    int gid = blockIdx.x * blockDim.x + threadIdx.x;
    if (gid >= BATCH * NTOT) return;
    int b = gid / NTOT;
    int n = gid - b * NTOT;

    const float* src; int S, off, scale;
    if (n < 19200)      { src = p0; S = 80; off = 0;     scale = 0; }
    else if (n < 24000) { src = p1; S = 40; off = 19200; scale = 1; }
    else                { src = p2; S = 20; off = 24000; scale = 2; }

    int m    = n - off;
    int a    = m % 3;
    int cell = m / 3;
    int gy   = cell / S;
    int gx   = cell - gy * S;
    float stride = (float)(8 << scale);

    const float* q = src + ((size_t)((b * S + gy) * S + gx)) * 255 + a * 85;

    float bx  = sigmoidf_(q[0]);
    float by  = sigmoidf_(q[1]);
    float bw  = __expf(fminf(q[2], 8.0f));
    float bh  = __expf(fminf(q[3], 8.0f));
    float obj = sigmoidf_(q[4]);

    float cx = (bx + (float)gx) * stride;
    float cy = (by + (float)gy) * stride;
    int   ai = scale * 3 + a;
    float pw = c_ancw[ai] * bw;   // (anc/stride)*bw*stride == anc*bw
    float ph = c_anch[ai] * bh;

    float x1 = fminf(fmaxf(cx - 0.5f * pw, 0.f), 639.f);
    float y1 = fminf(fmaxf(cy - 0.5f * ph, 0.f), 639.f);
    float x2 = fminf(fmaxf(cx + 0.5f * pw, 0.f), 639.f);
    float y2 = fminf(fmaxf(cy + 0.5f * ph, 0.f), 639.f);

    float* bo = boxes + (size_t)gid * 4;
    bo[0] = x1; bo[1] = y1; bo[2] = x2; bo[3] = y2;

    size_t sb = (size_t)b * NUMC * NTOT;
    #pragma unroll 4
    for (int c = 0; c < NUMC; ++c)
        scores[sb + (size_t)c * NTOT + n] = obj * sigmoidf_(q[5 + c]);
}

// ---------------------------------------------------------------------------
// Kernel 2: per (b,c) top-300 selection, fully LDS-resident (100.8 KB in the
// 320 KB CDNA5 LDS). 640 workgroups x 256 threads.
// ---------------------------------------------------------------------------
__global__ void yolo_topk(const float* __restrict__ scores,
                          const float* __restrict__ boxes,
                          float* __restrict__ sck,
                          int*   __restrict__ idxk,
                          float* __restrict__ boxesk)
{
    extern __shared__ unsigned char smem_raw[];
    float* s_sc = (float*)smem_raw;        // NTOT floats
    float* rv   = s_sc + NTOT;             // 256
    int*   ri   = (int*)(rv + 256);        // 256

    int bc  = blockIdx.x;
    int b   = bc / NUMC;
    int tid = threadIdx.x;

    const float* src = scores + (size_t)bc * NTOT;
#ifdef HAVE_ASYNC_LDS
    async_stage(src, s_sc, NTOT / 4, tid, 256);
#else
    for (int n = tid; n < NTOT; n += 256) s_sc[n] = src[n];
#endif
    __syncthreads();

    for (int t = 0; t < KTOP; ++t) {
        float bv = -3e38f; int bi = 0;
        for (int n = tid; n < NTOT; n += 256) {
            float v = s_sc[n];
            if (v > bv) { bv = v; bi = n; }     // strict > keeps lowest index on ties
        }
        rv[tid] = bv; ri[tid] = bi;
        __syncthreads();
        for (int s = 128; s > 0; s >>= 1) {
            if (tid < s) {
                float ov = rv[tid + s]; int oi = ri[tid + s];
                if (ov > rv[tid] || (ov == rv[tid] && oi < ri[tid])) { rv[tid] = ov; ri[tid] = oi; }
            }
            __syncthreads();
        }
        if (tid == 0) {
            sck [(size_t)bc * KTOP + t] = rv[0];
            idxk[(size_t)bc * KTOP + t] = ri[0];
            s_sc[ri[0]] = -3e38f;
        }
        __syncthreads();
    }

    for (int k = tid; k < KTOP; k += 256) {
        int id = idxk[(size_t)bc * KTOP + k];
        const float* bp = boxes  + ((size_t)b  * NTOT + id) * 4;
        float*       dp = boxesk + ((size_t)bc * KTOP + k ) * 4;
        dp[0] = bp[0]; dp[1] = bp[1]; dp[2] = bp[2]; dp[3] = bp[3];
    }
}

// ---------------------------------------------------------------------------
// Kernel 3: per (b,c) NMS. Build 304x304 suppression bitmask in parallel;
// the union's rank-2 term (area_i + area_j) runs on the WMMA unit.
// IoU threshold test is division-free: inter > t*(union+eps).
// 640 workgroups x 320 threads (10 wave32).
// ---------------------------------------------------------------------------
__global__ void yolo_nms(const float* __restrict__ sck,
                         const float* __restrict__ boxesk,
                         float* __restrict__ kept)
{
    __shared__ float    s_box [KPAD * 4];
    __shared__ float    s_area[KPAD];
    __shared__ unsigned s_mask[KPAD * 10];
    __shared__ unsigned s_valid[KPAD];
    __shared__ unsigned s_keep [KPAD];

    int bc  = blockIdx.x;
    int tid = threadIdx.x;

    for (int i = tid; i < KPAD; i += 320) {
        float x1 = 0.f, y1 = 0.f, x2 = 0.f, y2 = 0.f; unsigned v = 0;
        if (i < KTOP) {
            const float* bp = boxesk + ((size_t)bc * KTOP + i) * 4;
            x1 = bp[0]; y1 = bp[1]; x2 = bp[2]; y2 = bp[3];
            v  = (sck[(size_t)bc * KTOP + i] > CONF_T) ? 1u : 0u;
        }
        s_box[i * 4 + 0] = x1; s_box[i * 4 + 1] = y1;
        s_box[i * 4 + 2] = x2; s_box[i * 4 + 3] = y2;
        s_area[i]  = (x2 - x1) * (y2 - y1);
        s_valid[i] = v;
        s_keep[i]  = 0u;
    }
    for (int i = tid; i < KPAD * 10; i += 320) s_mask[i] = 0u;
    __syncthreads();

    int wave = tid >> 5;
    int lane = tid & 31;
    bool lo  = lane < 16;
    int  lm  = lane & 15;

    // Wave-uniform tile loop: EXEC is all-1s at every WMMA.
    for (int t = wave; t < NTILE * NTILE; t += 10) {
        int ti = t / NTILE, tj = t - ti * NTILE;
        int mb = ti * 16,   nb = tj * 16;

        float arow = lo ? s_area[mb + lm] : 0.0f;
        float acol = lo ? s_area[nb + lm] : 0.0f;

        v8f cz = {0.f,0.f,0.f,0.f,0.f,0.f,0.f,0.f};
        v8f d;
#ifdef HAVE_WMMA_F32X4
        // A (16x4): col0 = area_i, col1 = 1.  B (4x16): row0 = 1, row1 = area_j.
        // D[m][n] = area_i[m] + area_j[n]
        v2f af, bf;
        af.x = arow;             af.y = lo ? 1.0f : 0.0f;
        bf.x = lo ? 1.0f : 0.0f; bf.y = acol;
        d = __builtin_amdgcn_wmma_f32_16x16x4_f32(false, af, false, bf,
                                                  (short)0, cz, false, false);
#else
        // f16 fallback: scale by 1/256 so areas (<= 640*640) fit in f16 range.
        v16h ah, bh;
        #pragma unroll
        for (int e = 0; e < 16; ++e) { ah[e] = (_Float16)0.f; bh[e] = (_Float16)0.f; }
        ah[0] = (_Float16)(arow * (1.0f / 256.0f));
        ah[1] = (_Float16)(lo ? 256.0f : 0.0f);
        bh[0] = (_Float16)(lo ? 256.0f : 0.0f);
        bh[1] = (_Float16)(acol * (1.0f / 256.0f));
        d = __builtin_amdgcn_wmma_f32_16x16x32_f16(false, ah, false, bh,
                                                   (short)0, cz, false, false);
#endif
        int n = nb + (lo ? lane : lane - 16);
        float cx1 = s_box[n * 4 + 0], cy1 = s_box[n * 4 + 1];
        float cx2 = s_box[n * 4 + 2], cy2 = s_box[n * 4 + 3];

        #pragma unroll
        for (int r = 0; r < 8; ++r) {
            int mm = mb + (lo ? r : 8 + r);
            float rx1 = s_box[mm * 4 + 0], ry1 = s_box[mm * 4 + 1];
            float rx2 = s_box[mm * 4 + 2], ry2 = s_box[mm * 4 + 3];
            float ix1 = fmaxf(rx1, cx1), iy1 = fmaxf(ry1, cy1);
            float ix2 = fminf(rx2, cx2), iy2 = fminf(ry2, cy2);
            float inter = fmaxf(ix2 - ix1, 0.f) * fmaxf(iy2 - iy1, 0.f);
            float uni   = d[r] - inter;                 // area_m + area_n - inter
            // iou > t  <=>  inter > t*(uni+eps)   (uni+eps > 0 always)
            if (inter > IOU_T * (uni + 1e-9f))
                atomicOr(&s_mask[mm * 10 + (n >> 5)], 1u << (n & 31));
        }
    }
    __syncthreads();

    if (tid == 0) {
        unsigned supp[10] = {0,0,0,0,0,0,0,0,0,0};
        for (int i = 0; i < KTOP; ++i) {
            bool s = (supp[i >> 5] >> (i & 31)) & 1u;
            if (!s && s_valid[i]) {
                s_keep[i] = 1u;
                #pragma unroll
                for (int w = 0; w < 10; ++w) supp[w] |= s_mask[i * 10 + w];
            }
        }
    }
    __syncthreads();

    for (int k = tid; k < KTOP; k += 320)
        kept[(size_t)bc * KTOP + k] = s_keep[k] ? sck[(size_t)bc * KTOP + k] : 0.0f;
}

// ---------------------------------------------------------------------------
// Kernel 4: per-batch final top-300 over C*K = 24000 kept scores (96 KB LDS).
// Output: fboxes [8,300,4] | fscores [8,300] | labels [8,300] (as float).
// ---------------------------------------------------------------------------
__global__ void yolo_final(const float* __restrict__ kept,
                           const float* __restrict__ boxesk,
                           float* __restrict__ out)
{
    extern __shared__ unsigned char smem_raw[];
    float* s_sc = (float*)smem_raw;            // NUMC*KTOP floats
    float* rv   = s_sc + NUMC * KTOP;
    int*   ri   = (int*)(rv + 256);

    int b   = blockIdx.x;
    int tid = threadIdx.x;
    const int CK = NUMC * KTOP;

    const float* src = kept + (size_t)b * CK;
#ifdef HAVE_ASYNC_LDS
    async_stage(src, s_sc, CK / 4, tid, 256);
#else
    for (int n = tid; n < CK; n += 256) s_sc[n] = src[n];
#endif
    __syncthreads();

    float* fbox = out;
    float* fsc  = out + (size_t)BATCH * MAXDET * 4;
    float* flab = fsc + (size_t)BATCH * MAXDET;

    for (int t = 0; t < MAXDET; ++t) {
        float bv = -3e38f; int bi = 0;
        for (int n = tid; n < CK; n += 256) {
            float v = s_sc[n];
            if (v > bv) { bv = v; bi = n; }
        }
        rv[tid] = bv; ri[tid] = bi;
        __syncthreads();
        for (int s = 128; s > 0; s >>= 1) {
            if (tid < s) {
                float ov = rv[tid + s]; int oi = ri[tid + s];
                if (ov > rv[tid] || (ov == rv[tid] && oi < ri[tid])) { rv[tid] = ov; ri[tid] = oi; }
            }
            __syncthreads();
        }
        if (tid == 0) {
            int fi = ri[0];
            fsc [b * MAXDET + t] = rv[0];
            flab[b * MAXDET + t] = (float)(fi / KTOP);
            const float* bp = boxesk + ((size_t)b * CK + fi) * 4;
            float*       dp = fbox   + ((size_t)b * MAXDET + t) * 4;
            dp[0] = bp[0]; dp[1] = bp[1]; dp[2] = bp[2]; dp[3] = bp[3];
            s_sc[fi] = -3e38f;
        }
        __syncthreads();
    }
}

// ---------------------------------------------------------------------------
extern "C" void kernel_launch(void* const* d_in, const int* in_sizes, int n_in,
                              void* d_out, int out_size, void* d_ws, size_t ws_size,
                              hipStream_t stream)
{
    (void)in_sizes; (void)n_in; (void)out_size; (void)ws_size;
    const float* p0 = (const float*)d_in[0];
    const float* p1 = (const float*)d_in[1];
    const float* p2 = (const float*)d_in[2];

    float* ws = (float*)d_ws;
    size_t boxes_off  = 0;
    size_t scores_off = boxes_off  + (size_t)BATCH * NTOT * 4;
    size_t sck_off    = scores_off + (size_t)BATCH * NUMC * NTOT;
    size_t idx_off    = sck_off    + (size_t)BATCH * NUMC * KTOP;
    size_t boxesk_off = idx_off    + (size_t)BATCH * NUMC * KTOP;
    size_t kept_off   = boxesk_off + (size_t)BATCH * NUMC * KTOP * 4;

    float* boxes  = ws + boxes_off;
    float* scores = ws + scores_off;
    float* sck    = ws + sck_off;
    int*   idxk   = (int*)(ws + idx_off);
    float* boxesk = ws + boxesk_off;
    float* kept   = ws + kept_off;

    int total = BATCH * NTOT;
    yolo_decode<<<(total + 255) / 256, 256, 0, stream>>>(p0, p1, p2, boxes, scores);

    size_t sh2 = (size_t)(NTOT + 256) * sizeof(float) + 256 * sizeof(int);
    yolo_topk<<<BATCH * NUMC, 256, sh2, stream>>>(scores, boxes, sck, idxk, boxesk);

    yolo_nms<<<BATCH * NUMC, 320, 0, stream>>>(sck, boxesk, kept);

    size_t sh4 = (size_t)(NUMC * KTOP + 256) * sizeof(float) + 256 * sizeof(int);
    yolo_final<<<BATCH, 256, sh4, stream>>>(kept, boxesk, (float*)d_out);
}
